// CCN_2D_26113401160392
// MI455X (gfx1250) — matrix operand order repaired and checked by reading, not compiled, not run on hardware
//
#include <hip/hip_runtime.h>
#include <hip/hip_bf16.h>

#define NNODES 2000
#define KNB 16
#define CH 16
#define HID 16
#define KC_TOT 288   // 18 contractions * 16 channels = 9 * 32

typedef __attribute__((ext_vector_type(16))) _Float16 v16h;
typedef __attribute__((ext_vector_type(8)))  float    v8f;

#if defined(__has_builtin)
#if __has_builtin(__builtin_amdgcn_global_load_async_to_lds_b128) && \
    __has_builtin(__builtin_amdgcn_s_wait_asynccnt)
#define USE_ASYNC_LDS 1
// builtin expects pointers to 16B int vectors in AS1 (global) / AS3 (LDS)
typedef int v4i __attribute__((vector_size(16)));
typedef v4i __attribute__((address_space(1))) v4i_g;
typedef v4i __attribute__((address_space(3))) v4i_l;
#endif
#endif

// ---------------------------------------------------------------- gsum init
__global__ void init_gsum_kernel(float* gsum) {
    if (threadIdx.x < 64) gsum[threadIdx.x] = 0.f;
}

// ------------------------------------------------- neighbor list extraction
// One wave32 per node scans its adjacency row; ballot gives set columns in
// ascending order (matches jnp.sort of top_k indices for a 0/1 row).
__global__ void nbrs_kernel(const float* __restrict__ adj, int* __restrict__ nbrs) {
    int node = blockIdx.x * (blockDim.x >> 5) + (threadIdx.x >> 5);
    int lane = threadIdx.x & 31;
    if (node >= NNODES) return;
    const float* row = adj + (size_t)node * NNODES;
    int count = 0;
    for (int base = 0; base < NNODES; base += 32) {
        int col = base + lane;
        bool hit = (col < NNODES) && (row[col] > 0.5f);
        unsigned long long m = __ballot(hit);
        if (lane == 0) {
            while (m) {
                int b = __builtin_ctzll(m);
                m &= (m - 1ull);
                if (count < KNB) nbrs[node * KNB + count] = base + b;
                ++count;
            }
        }
    }
}

// ------------------------------------------------------- F0 + g0 channel sum
// F0[i,a,b,c] = (a==b) * X[nbrs[i,a], c]
__global__ void f0_kernel(const float* __restrict__ X, const int* __restrict__ nbrs,
                          float* __restrict__ F0, float* __restrict__ gsum) {
    __shared__ int   nb[KNB];
    __shared__ float red[256];
    int i = blockIdx.x, tid = threadIdx.x;
    if (tid < KNB) nb[tid] = nbrs[i * KNB + tid];
    __syncthreads();
    int a = tid >> 4, c = tid & 15;
    float x = X[(size_t)nb[a] * CH + c];
    float* base = F0 + ((size_t)i * 256 + a * 16) * 16;
    #pragma unroll
    for (int b = 0; b < 16; ++b) base[b * 16 + c] = (b == a) ? x : 0.f;
    red[tid] = x;
    __syncthreads();
    if (tid < 16) {
        float s = 0.f;
        for (int aa = 0; aa < 16; ++aa) s += red[aa * 16 + tid];
        atomicAdd(&gsum[tid], s);
    }
}

// -------------------------------------------------------------- layer step
// One workgroup (8 wave32s) per node. chi applied as an LDS gather (chi is a
// partial permutation); the dense 256x288 @ 288x16 GEMM runs on WMMA f16.
// Neighbor tiles stream into LDS via the CDNA5 async global->LDS path.
__global__ void __launch_bounds__(256)
step_kernel(const float* __restrict__ Fin, float* __restrict__ Fout,
            const int* __restrict__ nbrsG,
            const float* __restrict__ W, const float* __restrict__ bvec,
            float* __restrict__ gseg) {
    extern __shared__ char smem_raw[];
    float* S0 = (float*)smem_raw;       // generator blocks, 4096 f32 each
    float* S1 = S0 + 4096;
    float* S2 = S1 + 4096;
    float* D0 = S2 + 4096;
    float* D1 = D0 + 4096;
    float* D2 = D1 + 4096;
    float* V0 = D2 + 4096;              // 256 f32 each
    float* V1 = V0 + 256;
    float* V2 = V1 + 256;
    float* Fm = V2 + 256;               // neighbor tile, 4096 f32 (16B aligned)
    float* T  = Fm + 4096;              // promoted tile, 4096 f32
    float* biasL  = T + 4096;           // 16 f32
    int*   nbrs_i = (int*)(biasL + 16); // 16
    int*   nbrsnb = nbrs_i + 16;        // 256
    int*   pmap   = nbrsnb + 256;       // 256
    _Float16* Wl  = (_Float16*)(pmap + 256); // 4608 f16 (W[h][k])

    const int node = blockIdx.x;
    const int tid  = threadIdx.x;

    if (tid < KNB) nbrs_i[tid] = nbrsG[node * KNB + tid];
    if (tid < HID) biasL[tid]  = bvec[tid];
    __syncthreads();
    nbrsnb[tid] = nbrsG[nbrs_i[tid >> 4] * KNB + (tid & 15)];
    for (int u = tid; u < KC_TOT * HID; u += 256) Wl[u] = (_Float16)W[u];
    for (int u = tid; u < 4096; u += 256) S0[u] = 0.f;
    V1[tid] = 0.f; V2[tid] = 0.f;
    __syncthreads();

    {   // pmap[m][a]: position of nbrs_i[a] inside neighbor list of nbrs_i[m]
        int m = tid >> 4, a = tid & 15;
        int target = nbrs_i[a], pm = -1;
        #pragma unroll
        for (int p = 0; p < KNB; ++p)
            if (nbrsnb[m * 16 + p] == target) pm = p;
        pmap[m * 16 + a] = pm;
    }
    __syncthreads();

    const int a = tid >> 4, b = tid & 15;
    for (int m = 0; m < KNB; ++m) {
        const float* src = Fin + (size_t)nbrs_i[m] * 4096;
#if USE_ASYNC_LDS
        // async global->LDS DMA: no VGPR staging, tracked by ASYNCcnt
        #pragma unroll
        for (int it = 0; it < 4; ++it) {
            int u = tid + 256 * it;                    // float4 chunk index
            __builtin_amdgcn_global_load_async_to_lds_b128(
                (v4i_g*)(src + u * 4),
                (v4i_l*)(Fm + u * 4),
                0, 0);
        }
        __builtin_amdgcn_s_wait_asynccnt(0);
        __syncthreads();
#else
        {
            const float4* s4 = (const float4*)src;
            float4* d4 = (float4*)Fm;
            for (int u = tid; u < 1024; u += 256) d4[u] = s4[u];
        }
        __syncthreads();
#endif
        // T[a][b][c] = chi_m F chi_m^T  == gather through the permutation map
        int pa = pmap[m * 16 + a], pb = pmap[m * 16 + b];
        float* Trow = &T[(a * 16 + b) * 16];
        if (pa >= 0 && pb >= 0) {
            const float* s = &Fm[(pa * 16 + pb) * 16];
            #pragma unroll
            for (int c = 0; c < 16; ++c) Trow[c] = s[c];
        } else {
            #pragma unroll
            for (int c = 0; c < 16; ++c) Trow[c] = 0.f;
        }
        __syncthreads();
        // contraction generators; thread roles (r = output idx, c = channel)
        const int r = a, c = b;
        float s1 = 0.f, s2 = 0.f;
        #pragma unroll
        for (int x = 0; x < 16; ++x) {
            s1 += T[(x * 16 + r) * 16 + c];   // sum over a (b = r fixed)
            s2 += T[(r * 16 + x) * 16 + c];   // sum over b (a = r fixed)
        }
        S1[(m * 16 + r) * 16 + c] = s1;  V2[r * 16 + c] += s1;
        S2[(m * 16 + r) * 16 + c] = s2;  V1[r * 16 + c] += s2;
        D0[(m * 16 + r) * 16 + c] = T[(m * 16 + r) * 16 + c];
        D1[(m * 16 + r) * 16 + c] = T[(r * 16 + m) * 16 + c];
        D2[(m * 16 + r) * 16 + c] = T[(r * 16 + r) * 16 + c];
        #pragma unroll
        for (int ch = 0; ch < 16; ++ch)
            S0[(a * 16 + b) * 16 + ch] += T[(a * 16 + b) * 16 + ch];
        __syncthreads();
        if (tid < 16) {
            float v = 0.f;
            for (int x = 0; x < 16; ++x) v += S2[(m * 16 + x) * 16 + tid];
            V0[m * 16 + tid] = v;
        }
        __syncthreads();
    }

    // ---------------- dense GEMM: (16 rows b', fixed a'=rt) x W^T via WMMA
    const int wave = tid >> 5;
    const int lane = tid & 31;
    const int half = lane >> 4;   // wave32: lanes 0-15 vs 16-31
    const int ln   = lane & 15;
    for (int rt = wave; rt < 16; rt += 8) {
        v8f acc = {};
        #pragma unroll
        for (int kc = 0; kc < 9; ++kc) {
            v16h afrag, bfrag;
            #pragma unroll
            for (int j = 0; j < 8; ++j) {
                #pragma unroll
                for (int e = 0; e < 2; ++e) {
                    // 16-bit A 16x32 VGPR layout (ISA 7.12.2)
                    int kk = ((j < 4) ? (2 * j + e) : (16 + 2 * (j - 4) + e)) + 8 * half;
                    int k  = kc * 32 + kk;
                    int t  = k >> 4, ch = k & 15;
                    float val;
                    if (t < 12) {
                        const float* blk = S0 + (t >> 1) * 4096;  // S0,S1,S2,D0,D1,D2
                        int x = (t & 1) ? ln : rt;                // odd t = transposed
                        int y = (t & 1) ? rt : ln;
                        val = blk[(x * 16 + y) * 16 + ch];
                    } else {
                        const float* blk = V0 + ((t - 12) >> 1) * 256; // V0,V1,V2
                        int x = (t & 1) ? ln : rt;
                        val = blk[x * 16 + ch];
                    }
                    afrag[2 * j + e] = (_Float16)val;
                    // 16-bit B 32x16 layout: same K order as A
                    int kb = kc * 32 + kk;
                    bfrag[2 * j + e] = Wl[ln * KC_TOT + kb];      // B[k][h] = W[h][k]
                }
            }
            acc = __builtin_amdgcn_wmma_f32_16x16x32_f16(
                false, afrag, false, bfrag, (short)0, acc, false, false);
        }
        // epilogue: D[M=j+8*half][N=ln]; bias + ReLU + store + channel sum
        float bias = biasL[ln];
        float part = 0.f;
        #pragma unroll
        for (int j = 0; j < 8; ++j) {
            int bp = j + 8 * half;
            float v = acc[j] + bias;
            v = v > 0.f ? v : 0.f;
            if (Fout) Fout[((size_t)node * 256 + rt * 16 + bp) * 16 + ln] = v;
            part += v;
        }
        atomicAdd(&gseg[ln], part);
    }
}

// ---------------------------------------------------------------- finalize
__global__ void final_kernel(const float* __restrict__ gsum,
                             const float* __restrict__ fcw,
                             const float* __restrict__ fcb,
                             float* __restrict__ out) {
    if (threadIdx.x == 0) {
        float s = fcb[0];
        for (int j = 0; j < 48; ++j) s += gsum[j] * fcw[j];
        out[0] = s;
    }
}

extern "C" void kernel_launch(void* const* d_in, const int* in_sizes, int n_in,
                              void* d_out, int out_size, void* d_ws, size_t ws_size,
                              hipStream_t stream) {
    const float* X   = (const float*)d_in[0];
    const float* adj = (const float*)d_in[1];
    const float* W1  = (const float*)d_in[2];
    const float* b1  = (const float*)d_in[3];
    const float* W2  = (const float*)d_in[4];
    const float* b2  = (const float*)d_in[5];
    const float* fcw = (const float*)d_in[6];
    const float* fcb = (const float*)d_in[7];
    float* out = (float*)d_out;

    char* ws = (char*)d_ws;
    size_t off = 0;
    int* nbrs = (int*)(ws + off);
    off += ((size_t)NNODES * KNB * sizeof(int) + 255) & ~(size_t)255;
    float* gsum = (float*)(ws + off);
    off += 256;
    const size_t FB = (size_t)NNODES * 256 * 16 * sizeof(float);
    float* F0 = (float*)(ws + off); off += FB;
    float* F1 = (float*)(ws + off); off += FB;
    float* F2 = (off + FB <= ws_size) ? (float*)(ws + off) : nullptr;

    // dynamic LDS: 33552 f32 + 528 i32 + 4608 f16 = 145,536 B (< 160 KB)
    size_t smem = (size_t)(6 * 4096 + 3 * 256 + 4096 + 4096 + 16) * 4
                + (size_t)(16 + 256 + 256) * 4
                + (size_t)4608 * 2;

    hipLaunchKernelGGL(init_gsum_kernel, dim3(1), dim3(64), 0, stream, gsum);
    hipLaunchKernelGGL(nbrs_kernel, dim3(NNODES / 8), dim3(256), 0, stream, adj, nbrs);
    hipLaunchKernelGGL(f0_kernel, dim3(NNODES), dim3(256), 0, stream, X, nbrs, F0, gsum);
    hipLaunchKernelGGL(step_kernel, dim3(NNODES), dim3(256), smem, stream,
                       F0, F1, nbrs, W1, b1, gsum + 16);
    hipLaunchKernelGGL(step_kernel, dim3(NNODES), dim3(256), smem, stream,
                       F1, F2, nbrs, W2, b2, gsum + 32);
    hipLaunchKernelGGL(final_kernel, dim3(1), dim3(64), 0, stream, gsum, fcw, fcb, out);
}
